// GRULayer_45157286150493
// MI455X (gfx1250) — matrix operand order repaired and checked by reading, not compiled, not run on hardware
//
#include <hip/hip_runtime.h>

// ---------------------------------------------------------------- constants
static constexpr int Bb = 64;     // batch
static constexpr int Tt = 2048;   // time
static constexpr int Ii = 256;    // input dim
static constexpr int Hh = 256;    // hidden dim
static constexpr int N3 = 768;    // r|z|c combined width

typedef __bf16 bf16x8  __attribute__((ext_vector_type(8)));
typedef __bf16 bf16x16 __attribute__((ext_vector_type(16)));
typedef float  f32x8   __attribute__((ext_vector_type(8)));

// ---------------------------------------------------------------- helpers
__device__ inline unsigned short f2bf(float f) {
    unsigned u = __float_as_uint(f);
    unsigned r = (u + 0x7FFFu + ((u >> 16) & 1u)) >> 16;   // round-nearest-even
    return (unsigned short)r;
}

// A-fragment: two 8-half chunks (K {0-7,16-23} lanes 0-15 / {8-15,24-31} lanes 16-31)
__device__ inline bf16x16 ldfrag2(const unsigned short* p0, const unsigned short* p1) {
    union { bf16x16 v; bf16x8 h[2]; } u;
    u.h[0] = *(const bf16x8*)p0;
    u.h[1] = *(const bf16x8*)p1;
    return u.v;
}
// B-fragment: two 8-half chunks at p and p+16 (column of W stored transposed)
__device__ inline bf16x16 ldfrag16(const unsigned short* p) {
    union { bf16x16 v; bf16x8 h[2]; } u;
    u.h[0] = *(const bf16x8*)p;
    u.h[1] = *(const bf16x8*)(p + 16);
    return u.v;
}

__device__ inline f32x8 wmma_bf16(bf16x16 a, bf16x16 b, f32x8 c) {
    return __builtin_amdgcn_wmma_f32_16x16x32_bf16(false, a, false, b, (short)0, c,
                                                   false, false);
}

__device__ inline float sigmoidf_(float x) { return 1.0f / (1.0f + __expf(-x)); }

// monotonic-counter grid barrier (8 resident workgroups)
__device__ inline void grid_barrier(int* bar, int nblocks, int& gen) {
    ++gen;
    __threadfence();
    __syncthreads();
    if (threadIdx.x == 0) {
        __hip_atomic_fetch_add(bar, 1, __ATOMIC_RELEASE, __HIP_MEMORY_SCOPE_AGENT);
        while (__hip_atomic_load(bar, __ATOMIC_ACQUIRE, __HIP_MEMORY_SCOPE_AGENT) <
               nblocks * gen)
            __builtin_amdgcn_s_sleep(1);
    }
    __syncthreads();
}

// ---------------------------------------------------------------- prep: bf16 casts,
// transposed weight copies, combined bias, state init, barrier reset
__global__ __launch_bounds__(256) void gru_prep(
    const float* __restrict__ x,
    const float* __restrict__ r_wx, const float* __restrict__ r_wh,
    const float* __restrict__ r_b,
    const float* __restrict__ z_wx, const float* __restrict__ z_wh,
    const float* __restrict__ z_b,
    const float* __restrict__ input_w, const float* __restrict__ hidden_w,
    const float* __restrict__ bias,
    unsigned short* __restrict__ xb, unsigned short* __restrict__ w3t,
    unsigned short* __restrict__ wrzT, unsigned short* __restrict__ wcT,
    float* __restrict__ bias3, float* __restrict__ h,
    unsigned short* __restrict__ hb, unsigned short* __restrict__ hg,
    float* __restrict__ zb, int* __restrict__ bar)
{
    const size_t nt  = (size_t)gridDim.x * blockDim.x;
    const size_t gid = (size_t)blockIdx.x * blockDim.x + threadIdx.x;

    for (size_t i = gid; i < (size_t)Bb * Tt * Ii; i += nt) xb[i] = f2bf(x[i]);

    for (size_t i = gid; i < (size_t)N3 * Ii; i += nt) {          // W3^T [768][256]
        int n = (int)(i >> 8), k = (int)(i & 255);
        float v = (n < 256) ? r_wx[k * 256 + n]
                : (n < 512) ? z_wx[k * 256 + (n - 256)]
                            : input_w[k * 256 + (n - 512)];
        w3t[n * 256 + k] = f2bf(v);
    }
    for (size_t i = gid; i < (size_t)512 * 256; i += nt) {        // [r_wh|z_wh]^T
        int n = (int)(i >> 8), k = (int)(i & 255);
        float v = (n < 256) ? r_wh[k * 256 + n] : z_wh[k * 256 + (n - 256)];
        wrzT[n * 256 + k] = f2bf(v);
    }
    for (size_t i = gid; i < (size_t)256 * 256; i += nt) {        // hidden_w^T
        int n = (int)(i >> 8), k = (int)(i & 255);
        wcT[n * 256 + k] = f2bf(hidden_w[k * 256 + n]);
    }
    for (size_t i = gid; i < (size_t)N3; i += nt)
        bias3[i] = (i < 256) ? r_b[i] : (i < 512) ? z_b[i - 256] : bias[i - 512];

    for (size_t i = gid; i < (size_t)Bb * Hh; i += nt) {
        h[i] = 0.0f; hb[i] = 0; hg[i] = 0; zb[i] = 0.0f;
    }
    if (gid == 0) bar[0] = 0;
}

// ---------------------------------------------------------------- big input GEMM:
// Pre[t][b][0:768] = x[b,t,:] @ [r_wx|z_wx|input_w] + bias3   (M=131072,K=256,N=768)
__global__ __launch_bounds__(256) void gru_xproj(
    const unsigned short* __restrict__ xb,
    const unsigned short* __restrict__ w3t,
    const float* __restrict__ bias3,
    float* __restrict__ pre)
{
    __shared__ unsigned short As[128 * 64];   // 16 KB
    __shared__ unsigned short Bs[128 * 64];   // 16 KB

    const int tid  = threadIdx.x;
    const int w    = tid >> 5, lane = tid & 31;
    const int l16  = lane & 15, lhi = lane >> 4;
    const int Mb   = blockIdx.x * 128;
    const int Nb   = blockIdx.y * 128;
    const int wr   = w >> 1, wc = w & 1;       // wave grid 4x2, each 2x4 tiles

    f32x8 acc[2][4];
    const f32x8 zero = {0.f, 0.f, 0.f, 0.f, 0.f, 0.f, 0.f, 0.f};
#pragma unroll
    for (int i = 0; i < 2; ++i)
#pragma unroll
        for (int j = 0; j < 4; ++j) acc[i][j] = zero;

    const int r   = tid >> 1;                  // 0..127 row of LDS tile
    const int k0c = (tid & 1) * 32;            // half-row

    for (int kc0 = 0; kc0 < 256; kc0 += 64) {
        __syncthreads();
#pragma unroll
        for (int c = 0; c < 4; ++c) {
            *(uint4*)(As + r * 64 + k0c + c * 8) =
                *(const uint4*)(xb + (size_t)(Mb + r) * 256 + kc0 + k0c + c * 8);
            *(uint4*)(Bs + r * 64 + k0c + c * 8) =
                *(const uint4*)(w3t + (size_t)(Nb + r) * 256 + kc0 + k0c + c * 8);
        }
        __syncthreads();
#pragma unroll
        for (int kk = 0; kk < 64; kk += 32) {
            bf16x16 a[2], b[4];
#pragma unroll
            for (int i = 0; i < 2; ++i) {
                const unsigned short* ap =
                    As + (wr * 32 + i * 16 + l16) * 64 + kk + lhi * 8;
                a[i] = ldfrag2(ap, ap + 16);
            }
#pragma unroll
            for (int j = 0; j < 4; ++j)
                b[j] = ldfrag16(Bs + (wc * 64 + j * 16 + l16) * 64 + kk + lhi * 8);
#pragma unroll
            for (int i = 0; i < 2; ++i)
#pragma unroll
                for (int j = 0; j < 4; ++j)
                    acc[i][j] = wmma_bf16(a[i], b[j], acc[i][j]);
        }
    }
    // fused epilogue: scatter rows m=(b*T+t) -> Pre[t][b][n] (+bias)
#pragma unroll
    for (int i = 0; i < 2; ++i)
#pragma unroll
        for (int j = 0; j < 4; ++j) {
            const int n  = Nb + wc * 64 + j * 16 + l16;
            const float bv = bias3[n];
#pragma unroll
            for (int v = 0; v < 8; ++v) {
                const int m  = Mb + wr * 32 + i * 16 + v + (lhi << 3);
                const int bi = m >> 11;          // /T
                const int tt = m & 2047;         // %T
                pre[((size_t)tt * Bb + bi) * N3 + n] = acc[i][j][v] + bv;
            }
        }
}

// ---------------------------------------------------------------- persistent scan:
// 8 WGs x 8 waves; weight fragments REGISTER-resident for all 2048 steps;
// no LDS in the hot loop; 2 grid barriers per timestep.
__global__ __launch_bounds__(256) void gru_scan(
    const float* __restrict__ pre,
    const unsigned short* __restrict__ wrzT,
    const unsigned short* __restrict__ wcT,
    float* __restrict__ h, unsigned short* __restrict__ hb,
    unsigned short* __restrict__ hg, float* __restrict__ zb,
    float* __restrict__ out, int* __restrict__ bar)
{
    const int g    = blockIdx.x;               // 0..7
    const int tid  = threadIdx.x;
    const int w    = tid >> 5, lane = tid & 31;
    const int l16  = lane & 15, lhi = lane >> 4;

    // phase A: O1 = h @ [r_wh|z_wh]  -> 32 n-tiles x 4 m-tiles, 2 tiles/wave
    const int tnA  = (g * 8 + w) >> 1;                 // global n-tile 0..31
    const int nA0  = tnA * 16;                         // col base in [0,512)
    const int tmA0 = (w & 1) * 2;
    // phase B: O2 = (h*r) @ hidden_w -> 16 n-tiles x 4 m-tiles, 1 tile/wave
    const int tnB  = (g * 8 + w) >> 2;                 // 0..15
    const int nB0  = tnB * 16;
    const int tmB  = w & 3;

    // ---- load this wave's weight fragments ONCE; reuse for all 2048 steps
    bf16x16 wA[8];   // 64 VGPRs: 16 cols of [r_wh|z_wh]^T, all K
    bf16x16 wB[8];   // 64 VGPRs: 16 cols of hidden_w^T,   all K
#pragma unroll
    for (int k8 = 0; k8 < 8; ++k8)
        wA[k8] = ldfrag16(wrzT + (size_t)(nA0 + l16) * 256 + k8 * 32 + lhi * 8);
#pragma unroll
    for (int k8 = 0; k8 < 8; ++k8)
        wB[k8] = ldfrag16(wcT + (size_t)(nB0 + l16) * 256 + k8 * 32 + lhi * 8);

    int gen = 0;
    for (int t = 0; t < Tt; ++t) {
        const float* preT = pre + (size_t)t * Bb * N3;

        // ---- phase A GEMM (A-operand = h_bf16 from L2)
        f32x8 acc0 = {0.f,0.f,0.f,0.f,0.f,0.f,0.f,0.f};
        f32x8 acc1 = acc0;
#pragma unroll
        for (int k8 = 0; k8 < 8; ++k8) {
            const int kA = k8 * 32 + lhi * 8;
            const unsigned short* a0p = hb + (tmA0 * 16 + l16) * 256 + kA;
            const unsigned short* a1p = hb + ((tmA0 + 1) * 16 + l16) * 256 + kA;
            acc0 = wmma_bf16(ldfrag2(a0p, a0p + 16), wA[k8], acc0);
            acc1 = wmma_bf16(ldfrag2(a1p, a1p + 16), wA[k8], acc1);
        }
        // fused sigmoid epilogue (wave-uniform r/z branch):
        if (nA0 < 256) {          // r region -> hg = h*r (bf16)
#pragma unroll
            for (int q = 0; q < 2; ++q) {
                f32x8 a = q ? acc1 : acc0;
                const int bbase = (tmA0 + q) * 16;
#pragma unroll
                for (int v = 0; v < 8; ++v) {
                    const int bi = bbase + v + (lhi << 3);
                    const int n  = nA0 + l16;
                    const float s = sigmoidf_(a[v] + preT[bi * N3 + n]);
                    hg[bi * 256 + n] = f2bf(h[bi * 256 + n] * s);
                }
            }
        } else {                  // z region -> zb (f32)
#pragma unroll
            for (int q = 0; q < 2; ++q) {
                f32x8 a = q ? acc1 : acc0;
                const int bbase = (tmA0 + q) * 16;
#pragma unroll
                for (int v = 0; v < 8; ++v) {
                    const int bi = bbase + v + (lhi << 3);
                    const int n  = nA0 + l16;
                    zb[bi * 256 + (n - 256)] = sigmoidf_(a[v] + preT[bi * N3 + n]);
                }
            }
        }
        // pull next step's pre-activations toward L2/L0
        if (t + 1 < Tt) {
            const float* pn = pre + (size_t)(t + 1) * Bb * N3;
            __builtin_prefetch(pn + ((size_t)g * 256 + tid) * 24, 0, 1);
        }
        grid_barrier(bar, 8, gen);

        // ---- phase B GEMM (A-operand = hg_bf16 from L2)
        f32x8 acc = {0.f,0.f,0.f,0.f,0.f,0.f,0.f,0.f};
#pragma unroll
        for (int k8 = 0; k8 < 8; ++k8) {
            const unsigned short* ap = hg + (tmB * 16 + l16) * 256 + k8 * 32 + lhi * 8;
            acc = wmma_bf16(ldfrag2(ap, ap + 16), wB[k8], acc);
        }
        // fused tanh + gate blend + state/output writeback
#pragma unroll
        for (int v = 0; v < 8; ++v) {
            const int bi = tmB * 16 + v + (lhi << 3);
            const int n  = nB0 + l16;
            const float c  = tanhf(acc[v] + preT[bi * N3 + 512 + n]);
            const float zv = zb[bi * 256 + n];
            const float hn = zv * h[bi * 256 + n] + (1.0f - zv) * c;
            h [bi * 256 + n] = hn;
            hb[bi * 256 + n] = f2bf(hn);
            out[(size_t)bi * Tt * Hh + (size_t)t * Hh + n] = hn;
        }
        grid_barrier(bar, 8, gen);
    }
}

// ---------------------------------------------------------------- launcher
extern "C" void kernel_launch(void* const* d_in, const int* in_sizes, int n_in,
                              void* d_out, int out_size, void* d_ws, size_t ws_size,
                              hipStream_t stream) {
    const float* x        = (const float*)d_in[0];
    const float* r_wx     = (const float*)d_in[1];
    const float* r_wh     = (const float*)d_in[2];
    const float* r_b      = (const float*)d_in[3];
    const float* z_wx     = (const float*)d_in[4];
    const float* z_wh     = (const float*)d_in[5];
    const float* z_b      = (const float*)d_in[6];
    const float* input_w  = (const float*)d_in[7];
    const float* hidden_w = (const float*)d_in[8];
    const float* bias     = (const float*)d_in[9];
    float* out = (float*)d_out;

    char* ws = (char*)d_ws;
    size_t off = 0;
    auto take = [&off](size_t bytes) -> size_t {
        size_t r = off;
        off += (bytes + 255) & ~(size_t)255;
        return r;
    };
    unsigned short* xb   = (unsigned short*)(ws + take((size_t)Bb * Tt * Ii * 2));
    unsigned short* w3t  = (unsigned short*)(ws + take((size_t)N3 * Ii * 2));
    unsigned short* wrzT = (unsigned short*)(ws + take((size_t)512 * 256 * 2));
    unsigned short* wcT  = (unsigned short*)(ws + take((size_t)256 * 256 * 2));
    float*          b3   = (float*)(ws + take((size_t)N3 * 4));
    float*          pre  = (float*)(ws + take((size_t)Tt * Bb * N3 * 4));
    float*          h    = (float*)(ws + take((size_t)Bb * Hh * 4));
    unsigned short* hb   = (unsigned short*)(ws + take((size_t)Bb * Hh * 2));
    unsigned short* hg   = (unsigned short*)(ws + take((size_t)Bb * Hh * 2));
    float*          zb   = (float*)(ws + take((size_t)Bb * Hh * 4));
    int*            bar  = (int*)(ws + take(256));

    gru_prep<<<4096, 256, 0, stream>>>(x, r_wx, r_wh, r_b, z_wx, z_wh, z_b,
                                       input_w, hidden_w, bias,
                                       xb, w3t, wrzT, wcT, b3, h, hb, hg, zb, bar);
    gru_xproj<<<dim3((Bb * Tt) / 128, N3 / 128, 1), 256, 0, stream>>>(xb, w3t, b3, pre);
    gru_scan<<<8, 256, 0, stream>>>(pre, wrzT, wcT, h, hb, hg, zb, out, bar);
}
